// MultiHeadSelfAttention_18073222382060
// MI455X (gfx1250) — matrix operand order repaired and checked by reading, not compiled, not run on hardware
//
#include <hip/hip_runtime.h>
#include <hip/hip_bf16.h>
#include <stdint.h>

#define E_DIM 1024
#define H_DIM 16
#define D_DIM 64
#define S_LEN 2048
#define B_SZ  2

typedef __attribute__((ext_vector_type(16))) __bf16 bf16x16;
typedef __attribute__((ext_vector_type(8)))  float  f32x8;

union FragU {
  bf16x16 v;
  uint4 q[2];
  unsigned short s[16];
};

// ---- CDNA5 async global->LDS copy (ASYNCcnt path), with safe fallback ----
// Probe-derived prototype: (v4i32 AS1*, v4i32 AS3*, imm offset, imm cpol)
typedef int v4i_t __attribute__((__vector_size__(16)));
typedef __attribute__((address_space(1))) v4i_t* gv4i_p;
typedef __attribute__((address_space(3))) v4i_t* lv4i_p;

#if __has_builtin(__builtin_amdgcn_global_load_async_to_lds_b128)
#define HAVE_ASYNC_LDS 1
#define ASYNC_CP16(gp, lp)                                                   \
  __builtin_amdgcn_global_load_async_to_lds_b128((gv4i_p)(gp), (lv4i_p)(lp), \
                                                 0, 0)
#else
#define HAVE_ASYNC_LDS 0
#define ASYNC_CP16(gp, lp)                                                   \
  do { *(uint4*)(lp) = *(const uint4*)(gp); } while (0)
#endif

__device__ __forceinline__ void async_commit_barrier() {
#if HAVE_ASYNC_LDS
#if __has_builtin(__builtin_amdgcn_s_wait_asynccnt)
  __builtin_amdgcn_s_wait_asynccnt(0);
#else
  asm volatile("s_wait_asynccnt 0" ::: "memory");
#endif
#endif
  __syncthreads();
}

__device__ __forceinline__ unsigned short f32_to_bf16(float f) {
  union { float f; uint32_t u; } a; a.f = f;
  uint32_t lsb = (a.u >> 16) & 1u;
  a.u += 0x7fffu + lsb;               // round-to-nearest-even
  return (unsigned short)(a.u >> 16);
}

__device__ __forceinline__ f32x8 zero8() {
  f32x8 z;
#pragma unroll
  for (int i = 0; i < 8; ++i) z[i] = 0.0f;
  return z;
}

// A-fragment 16x32 bf16: lane (L&15)=row M; lanes 0-15 hold
// K=0..15 contiguous; lanes 16-31 hold K=16..31 (matches B-fragment order).
__device__ __forceinline__ bf16x16 load_a_frag(const unsigned short* rowPtr, int hi) {
  FragU f;
  f.q[0] = *(const uint4*)(rowPtr + hi * 16);
  f.q[1] = *(const uint4*)(rowPtr + hi * 16 + 8);
  return f.v;
}

// B-fragment 32x16 bf16: lane (L&15)=column N; per-lane 16 contiguous K
// values starting at k0 + (L>>4)*16.
__device__ __forceinline__ bf16x16 load_b_frag(const unsigned short* p) {
  FragU f;
  f.q[0] = *(const uint4*)(p);
  f.q[1] = *(const uint4*)(p + 8);
  return f.v;
}

__device__ __forceinline__ f32x8 wmma_bf16(bf16x16 a, bf16x16 b, f32x8 c) {
  return __builtin_amdgcn_wmma_f32_16x16x32_bf16(false, a, false, b, (short)0, c, false, false);
}

// Software-pipelined 16x64-tile GEMM loop, unrolled x2 so the two fragment
// sets alternate without register moves: each set dies at its WMMA group and
// is refilled by fresh loads for k+64.
__device__ __forceinline__ void gemm_tile_k(const unsigned short* aPtr,
                                            const unsigned short* const bPtr[4],
                                            int hi, f32x8 acc[4]) {
  bf16x16 a0 = load_a_frag(aPtr, hi);
  bf16x16 b0[4];
#pragma unroll
  for (int f = 0; f < 4; ++f) b0[f] = load_b_frag(bPtr[f]);

  for (int k0 = 0; k0 < E_DIM - 64; k0 += 64) {
    __builtin_prefetch(aPtr + k0 + 128, 0, 1);     // global_prefetch_b8
    // stage 1: preload k0+32 into set1, consume set0 (k0)
    bf16x16 a1 = load_a_frag(aPtr + k0 + 32, hi);
    bf16x16 b1[4];
#pragma unroll
    for (int f = 0; f < 4; ++f) b1[f] = load_b_frag(bPtr[f] + k0 + 32);
#pragma unroll
    for (int f = 0; f < 4; ++f) acc[f] = wmma_bf16(a0, b0[f], acc[f]);
    // stage 2: preload k0+64 into set0, consume set1 (k0+32)
    a0 = load_a_frag(aPtr + k0 + 64, hi);
#pragma unroll
    for (int f = 0; f < 4; ++f) b0[f] = load_b_frag(bPtr[f] + k0 + 64);
#pragma unroll
    for (int f = 0; f < 4; ++f) acc[f] = wmma_bf16(a1, b1[f], acc[f]);
  }
  // tail: set0 holds k = E_DIM-64; last chunk at E_DIM-32
  {
    bf16x16 a1 = load_a_frag(aPtr + E_DIM - 32, hi);
    bf16x16 b1[4];
#pragma unroll
    for (int f = 0; f < 4; ++f) b1[f] = load_b_frag(bPtr[f] + E_DIM - 32);
#pragma unroll
    for (int f = 0; f < 4; ++f) acc[f] = wmma_bf16(a0, b0[f], acc[f]);
#pragma unroll
    for (int f = 0; f < 4; ++f) acc[f] = wmma_bf16(a1, b1[f], acc[f]);
  }
}

// ---------------------------------------------------------------- stage 0
__global__ void cvt_f32_bf16(const float* __restrict__ src,
                             unsigned short* __restrict__ dst, int n) {
  int i = blockIdx.x * blockDim.x + threadIdx.x;
  if (i < n) dst[i] = f32_to_bf16(src[i]);
}

// ---------------------------------------------------------------- stage 1
// QKV projection: y = X @ W^T + b.  One wave = 16x64 tile = one head slice.
// Q,K stored [B,H,S,D]; V stored transposed [B,H,D,S].
__global__ __launch_bounds__(256)
void qkv_kernel(const unsigned short* __restrict__ Xb,
                const unsigned short* __restrict__ Wqb,
                const unsigned short* __restrict__ Wkb,
                const unsigned short* __restrict__ Wvb,
                const float* __restrict__ bq,
                const float* __restrict__ bk,
                const float* __restrict__ bv,
                unsigned short* __restrict__ Qb,
                unsigned short* __restrict__ Kb,
                unsigned short* __restrict__ Vtb) {
  const int lane = threadIdx.x & 31;
  const int hi = lane >> 4, lo = lane & 15;
  const int wid = (blockIdx.x * blockDim.x + threadIdx.x) >> 5;
  const int NT = E_DIM / 64;            // 16 n-tiles (heads)
  const int MT = (B_SZ * S_LEN) / 16;   // 256 m-tiles
  const int mat = wid / (MT * NT);
  const int rem = wid % (MT * NT);
  const int mtile = rem / NT;
  const int ntile = rem % NT;

  const unsigned short* W = (mat == 0) ? Wqb : (mat == 1) ? Wkb : Wvb;
  const float* bias = (mat == 0) ? bq : (mat == 1) ? bk : bv;

  const unsigned short* aPtr = Xb + (size_t)(mtile * 16 + lo) * E_DIM;
  const unsigned short* bPtr[4];
#pragma unroll
  for (int f = 0; f < 4; ++f)
    bPtr[f] = W + (size_t)(ntile * 64 + f * 16 + lo) * E_DIM + hi * 16;

  f32x8 acc[4];
#pragma unroll
  for (int f = 0; f < 4; ++f) acc[f] = zero8();

  gemm_tile_k(aPtr, bPtr, hi, acc);

#pragma unroll
  for (int f = 0; f < 4; ++f) {
    const float bval = bias[ntile * 64 + f * 16 + lo];
    const int d = f * 16 + lo;
#pragma unroll
    for (int r = 0; r < 8; ++r) {
      unsigned short h16 = f32_to_bf16(acc[f][r] + bval);
      int m = mtile * 16 + r + 8 * hi;
      int bb = m >> 11;               // / S_LEN
      int s = m & (S_LEN - 1);
      if (mat == 0)
        Qb[((size_t)(bb * H_DIM + ntile) * S_LEN + s) * D_DIM + d] = h16;
      else if (mat == 1)
        Kb[((size_t)(bb * H_DIM + ntile) * S_LEN + s) * D_DIM + d] = h16;
      else
        Vtb[((size_t)(bb * H_DIM + ntile) * D_DIM + d) * S_LEN + s] = h16;
    }
  }
}

// ---------------------------------------------------------------- stage 2
// Flash attention.  All 8 waves of a block share (b,h) and stream the SAME
// 32-key blocks, so K/V tiles are staged in LDS once per block via
// double-buffered async global->LDS copies (ASYNCcnt), then consumed as
// WMMA B-fragments straight out of LDS.
__global__ __launch_bounds__(256)
void attn_kernel(const unsigned short* __restrict__ Qb,
                 const unsigned short* __restrict__ Kb,
                 const unsigned short* __restrict__ Vtb,
                 unsigned short* __restrict__ Cb) {
  __shared__ __align__(16) unsigned short Kbuf[2][32 * 64];  // [key][d]  4KB x2
  __shared__ __align__(16) unsigned short Vbuf[2][64 * 32];  // [d][s]    4KB x2
  __shared__ __align__(16) unsigned short Pbuf[8][16 * 32];  // per-wave P tile

  const int tid = threadIdx.x;
  const int lane = tid & 31;
  const int hi = lane >> 4, lo = lane & 15;
  const int wv = tid >> 5;
  const int wid = (blockIdx.x * blockDim.x + tid) >> 5;
  const int QT = S_LEN / 16;            // 128 query tiles per head
  const int qtile = wid % QT;
  const int h = (wid / QT) % H_DIM;
  const int b = wid / (QT * H_DIM);

  const unsigned short* Qh = Qb + (size_t)((b * H_DIM + h) * S_LEN) * D_DIM;
  const unsigned short* Kh = Kb + (size_t)((b * H_DIM + h) * S_LEN) * D_DIM;
  const unsigned short* Vth = Vtb + (size_t)((b * H_DIM + h) * D_DIM) * S_LEN;

  // cooperative K/V block copy: 256 threads x 16B each per matrix
  const int krow = tid >> 3, kchunk = (tid & 7) * 8;   // 32 rows x 128B
  const int vrow = tid >> 2, vchunk = (tid & 3) * 8;   // 64 rows x 64B

  const int qrow = qtile * 16 + lo;
  bf16x16 aq0 = load_a_frag(Qh + (size_t)qrow * D_DIM + 0, hi);
  bf16x16 aq1 = load_a_frag(Qh + (size_t)qrow * D_DIM + 32, hi);

  f32x8 acc[4];
#pragma unroll
  for (int f = 0; f < 4; ++f) acc[f] = zero8();
  float mrow[8], lrow[8];
#pragma unroll
  for (int r = 0; r < 8; ++r) { mrow[r] = -3.0e38f; lrow[r] = 0.0f; }

  const float scale = 0.125f;           // 1/sqrt(64)
  unsigned short* pbuf = Pbuf[wv];

  // prologue: stage block 0
  ASYNC_CP16(Kh + (size_t)krow * D_DIM + kchunk, &Kbuf[0][krow * 64 + kchunk]);
  ASYNC_CP16(Vth + (size_t)vrow * S_LEN + vchunk, &Vbuf[0][vrow * 32 + vchunk]);
  async_commit_barrier();

  int buf = 0;
  for (int kb = 0; kb < S_LEN; kb += 32) {
    const int nbuf = buf ^ 1;
    if (kb + 32 < S_LEN) {              // uniform: issue next block's copy
      const int kn = kb + 32;
      ASYNC_CP16(Kh + (size_t)(kn + krow) * D_DIM + kchunk,
                 &Kbuf[nbuf][krow * 64 + kchunk]);
      ASYNC_CP16(Vth + (size_t)vrow * S_LEN + kn + vchunk,
                 &Vbuf[nbuf][vrow * 32 + vchunk]);
    }

    const unsigned short* kB = &Kbuf[buf][0];
    const unsigned short* vB = &Vbuf[buf][0];

    // ---- scores S = Q @ K^T for this 16x32 key block, B-frags from LDS
    f32x8 s0 = zero8(), s1 = zero8();
    {
      const unsigned short* kp = kB + (lo) * 64 + hi * 16;
      s0 = wmma_bf16(aq0, load_b_frag(kp), s0);
      s0 = wmma_bf16(aq1, load_b_frag(kp + 32), s0);
    }
    {
      const unsigned short* kp = kB + (16 + lo) * 64 + hi * 16;
      s1 = wmma_bf16(aq0, load_b_frag(kp), s1);
      s1 = wmma_bf16(aq1, load_b_frag(kp + 32), s1);
    }

    // ---- online softmax; rows live across 16-lane groups
#pragma unroll
    for (int r = 0; r < 8; ++r) {
      float v0 = s0[r] * scale;
      float v1 = s1[r] * scale;
      float mx = fmaxf(v0, v1);
#pragma unroll
      for (int msk = 1; msk < 16; msk <<= 1)
        mx = fmaxf(mx, __shfl_xor(mx, msk, 32));
      float mnew = fmaxf(mrow[r], mx);
      float corr = __expf(mrow[r] - mnew);
      float p0 = __expf(v0 - mnew);
      float p1 = __expf(v1 - mnew);
      float rs = p0 + p1;
#pragma unroll
      for (int msk = 1; msk < 16; msk <<= 1)
        rs += __shfl_xor(rs, msk, 32);
      lrow[r] = lrow[r] * corr + rs;
      mrow[r] = mnew;
      acc[0][r] *= corr; acc[1][r] *= corr;
      acc[2][r] *= corr; acc[3][r] *= corr;
      s0[r] = p0; s1[r] = p1;
    }

    // ---- C-layout -> A-layout via per-wave LDS bounce
#pragma unroll
    for (int r = 0; r < 8; ++r) {
      int row = r + 8 * hi;
      pbuf[row * 32 + lo] = f32_to_bf16(s0[r]);
      pbuf[row * 32 + 16 + lo] = f32_to_bf16(s1[r]);
    }
    asm volatile("s_wait_dscnt 0" ::: "memory");   // cross-lane LDS RAW
    bf16x16 ap = load_a_frag(pbuf + lo * 32, hi);

    // ---- acc += P @ V, V B-frags from LDS ([d][s] layout)
#pragma unroll
    for (int dt = 0; dt < 4; ++dt) {
      const unsigned short* vp = vB + (dt * 16 + lo) * 32 + hi * 16;
      acc[dt] = wmma_bf16(ap, load_b_frag(vp), acc[dt]);
    }

    async_commit_barrier();   // my async writes landed; everyone done reading
    buf = nbuf;
  }

  // ---- normalize and write concat [B,S,E] (E index = h*64 + d), bf16
#pragma unroll
  for (int r = 0; r < 8; ++r) {
    float inv = 1.0f / lrow[r];
    int q = qtile * 16 + r + 8 * hi;
    unsigned short* cp = Cb + ((size_t)(b * S_LEN + q) * E_DIM) + h * D_DIM;
#pragma unroll
    for (int dt = 0; dt < 4; ++dt)
      cp[dt * 16 + lo] = f32_to_bf16(acc[dt][r] * inv);
  }
}

// ---------------------------------------------------------------- stage 3
// Output projection: out = concat @ Wo^T + bo, f32 output.
__global__ __launch_bounds__(256)
void out_kernel(const unsigned short* __restrict__ Cbuf,
                const unsigned short* __restrict__ Wob,
                const float* __restrict__ bo,
                float* __restrict__ out) {
  const int lane = threadIdx.x & 31;
  const int hi = lane >> 4, lo = lane & 15;
  const int wid = (blockIdx.x * blockDim.x + threadIdx.x) >> 5;
  const int NT = E_DIM / 64;            // 16
  const int mtile = wid / NT;
  const int ntile = wid % NT;

  const unsigned short* aPtr = Cbuf + (size_t)(mtile * 16 + lo) * E_DIM;
  const unsigned short* bPtr[4];
#pragma unroll
  for (int f = 0; f < 4; ++f)
    bPtr[f] = Wob + (size_t)(ntile * 64 + f * 16 + lo) * E_DIM + hi * 16;

  f32x8 acc[4];
#pragma unroll
  for (int f = 0; f < 4; ++f) acc[f] = zero8();

  gemm_tile_k(aPtr, bPtr, hi, acc);

#pragma unroll
  for (int f = 0; f < 4; ++f) {
    const float bval = bo[ntile * 64 + f * 16 + lo];
#pragma unroll
    for (int r = 0; r < 8; ++r) {
      int m = mtile * 16 + r + 8 * hi;
      out[(size_t)m * E_DIM + ntile * 64 + f * 16 + lo] = acc[f][r] + bval;
    }
  }
}

// ---------------------------------------------------------------- launch
extern "C" void kernel_launch(void* const* d_in, const int* in_sizes, int n_in,
                              void* d_out, int out_size, void* d_ws, size_t ws_size,
                              hipStream_t stream) {
  const float* x  = (const float*)d_in[0];
  const float* Wq = (const float*)d_in[1];
  const float* bq = (const float*)d_in[2];
  const float* Wk = (const float*)d_in[3];
  const float* bk = (const float*)d_in[4];
  const float* Wv = (const float*)d_in[5];
  const float* bv = (const float*)d_in[6];
  const float* Wo = (const float*)d_in[7];
  const float* bo = (const float*)d_in[8];
  float* out = (float*)d_out;

  unsigned short* ws = (unsigned short*)d_ws;
  const size_t nX = (size_t)B_SZ * S_LEN * E_DIM;   // 4,194,304
  const size_t nW = (size_t)E_DIM * E_DIM;          // 1,048,576
  unsigned short* Xb  = ws;
  unsigned short* Wqb = Xb + nX;
  unsigned short* Wkb = Wqb + nW;
  unsigned short* Wvb = Wkb + nW;
  unsigned short* Wob = Wvb + nW;
  unsigned short* Qb  = Wob + nW;
  unsigned short* Kb  = Qb + nX;
  unsigned short* Vtb = Kb + nX;
  unsigned short* Cb  = Vtb + nX;

  cvt_f32_bf16<<<(int)(nX / 256), 256, 0, stream>>>(x, Xb, (int)nX);
  cvt_f32_bf16<<<(int)(nW / 256), 256, 0, stream>>>(Wq, Wqb, (int)nW);
  cvt_f32_bf16<<<(int)(nW / 256), 256, 0, stream>>>(Wk, Wkb, (int)nW);
  cvt_f32_bf16<<<(int)(nW / 256), 256, 0, stream>>>(Wv, Wvb, (int)nW);
  cvt_f32_bf16<<<(int)(nW / 256), 256, 0, stream>>>(Wo, Wob, (int)nW);

  // 3 matrices x 256 m-tiles x 16 n-tiles = 12288 waves / 8 per block
  qkv_kernel<<<1536, 256, 0, stream>>>(Xb, Wqb, Wkb, Wvb, bq, bk, bv, Qb, Kb, Vtb);

  // 2*16*128 = 4096 waves / 8 per block
  attn_kernel<<<512, 256, 0, stream>>>(Qb, Kb, Vtb, Cb);

  // 256*16 = 4096 waves / 8 per block
  out_kernel<<<512, 256, 0, stream>>>(Cb, Wob, bo, out);
}
